// RNNPrediction_13855564497134
// MI455X (gfx1250) — matrix operand order repaired and checked by reading, not compiled, not run on hardware
//
#include <hip/hip_runtime.h>

typedef __attribute__((ext_vector_type(16))) __bf16 v16bf;
typedef __attribute__((ext_vector_type(8)))  float  v8f;
typedef int v4i_vs __attribute__((vector_size(16)));   // matches builtin param type

#define BB 4096
#define DD 512
#define HH 2048
#define NSTEP 16
#define LDP 40   // LDS row stride in halves: 80B, 16B aligned, bank-spread

#define __AS1 __attribute__((address_space(1)))
#define __AS3 __attribute__((address_space(3)))

#if __has_builtin(__builtin_amdgcn_global_load_async_to_lds_b128)
#define USE_ASYNC_LDS 1
#else
#define USE_ASYNC_LDS 0
#endif

__device__ __forceinline__ void wait_async0() {
#if __has_builtin(__builtin_amdgcn_s_wait_asynccnt)
    __builtin_amdgcn_s_wait_asynccnt(0);
#else
    asm volatile("s_wait_asynccnt 0x0" ::: "memory");
#endif
}

// ---------- helpers ----------
__device__ __forceinline__ unsigned short f2bf_u(float f) {
    union { float f; unsigned u; } c;
    c.f = f;
    unsigned u = c.u;
    return (unsigned short)((u + 0x7FFFu + ((u >> 16) & 1u)) >> 16);
}

// ---------- preprocessing kernels ----------
__global__ void k_transpose_to_bf16(const float* __restrict__ W,
                                    unsigned short* __restrict__ Wt,
                                    int Kdim, int Ndim) {
    size_t idx = (size_t)blockIdx.x * blockDim.x + threadIdx.x; // coalesced read
    int k = (int)(idx / Ndim);
    int n = (int)(idx % Ndim);
    Wt[(size_t)n * Kdim + k] = f2bf_u(W[idx]);
}

__global__ void k_convert_bf16(const float* __restrict__ in,
                               unsigned short* __restrict__ out) {
    size_t i = (size_t)blockIdx.x * blockDim.x + threadIdx.x;
    out[i] = f2bf_u(in[i]);
}

__global__ void k_zero_u32(unsigned int* __restrict__ p) {
    size_t i = (size_t)blockIdx.x * blockDim.x + threadIdx.x;
    p[i] = 0u;
}

// ---------- bf16 WMMA GEMM, double-buffered LDS, pipelined ----------
// C[M,N] = A[M,K](bf16) * Bt[N,K]^T(bf16)  (+bias[N]) (+addsrc[M,N]) (tanh?)
//   -> outF[M, ldout] (f32, optional)  and/or  outB[M,N] (bf16, optional)
// Requires M%128==0, N%128==0, K%32==0. Block = 256 threads (8 waves).
__global__ __launch_bounds__(256)
void k_gemm_bf16_wmma(const unsigned short* __restrict__ A,
                      const unsigned short* __restrict__ Bt,
                      const float* __restrict__ bias,
                      const float* __restrict__ addsrc,
                      float* __restrict__ outF, long long ldout,
                      unsigned short* __restrict__ outB,
                      int M, int N, int K, int dotanh) {
    __shared__ unsigned short As[2][128][LDP];
    __shared__ unsigned short Bs[2][128][LDP];

    const int t    = threadIdx.x;
    const int lane = t & 31;
    const int w    = t >> 5;       // wave 0..7
    const int wm   = w >> 2;       // 0..1  (64 rows each)
    const int wn   = w & 3;        // 0..3  (32 cols each)
    const int tileM = blockIdx.y * 128;
    const int tileN = blockIdx.x * 128;

    // this thread's two 16B staging chunks per matrix (512 chunks / 256 threads)
    const int r0 = t >> 2,            c0 = (t & 3) << 3;          // chunk 0
    const int r1 = (t + 256) >> 2,    c1 = (t & 3) << 3;          // chunk 1
    const unsigned short* gA0 = A  + (size_t)(tileM + r0) * K + c0;
    const unsigned short* gA1 = A  + (size_t)(tileM + r1) * K + c1;
    const unsigned short* gB0 = Bt + (size_t)(tileN + r0) * K + c0;
    const unsigned short* gB1 = Bt + (size_t)(tileN + r1) * K + c1;

#if !USE_ASYNC_LDS
    uint4 rg[4];
#endif

    const v8f vzero = {0.f, 0.f, 0.f, 0.f, 0.f, 0.f, 0.f, 0.f};
    v8f acc[4][2];
#pragma unroll
    for (int mi = 0; mi < 4; ++mi)
#pragma unroll
        for (int ni = 0; ni < 2; ++ni) acc[mi][ni] = vzero;

    union Frag { uint4 u[2]; v16bf v; };

    const int mrow  = lane & 15;
    const int koffA = (lane >> 4) << 3;     // 0 or 8   (A fragment layout)
    const int koffB = (lane >> 4) << 3;     // 0 or 8   (B fragment layout)

#if USE_ASYNC_LDS
    auto stage_async = [&](int buf, int kb) {
        __builtin_amdgcn_global_load_async_to_lds_b128(
            (__AS1 v4i_vs*)(gA0 + kb), (__AS3 v4i_vs*)&As[buf][r0][c0], 0, 0);
        __builtin_amdgcn_global_load_async_to_lds_b128(
            (__AS1 v4i_vs*)(gA1 + kb), (__AS3 v4i_vs*)&As[buf][r1][c1], 0, 0);
        __builtin_amdgcn_global_load_async_to_lds_b128(
            (__AS1 v4i_vs*)(gB0 + kb), (__AS3 v4i_vs*)&Bs[buf][r0][c0], 0, 0);
        __builtin_amdgcn_global_load_async_to_lds_b128(
            (__AS1 v4i_vs*)(gB1 + kb), (__AS3 v4i_vs*)&Bs[buf][r1][c1], 0, 0);
    };
    stage_async(0, 0);          // prologue: tile 0 -> buffer 0
    wait_async0();
    __syncthreads();
#else
    auto load_regs = [&](int kb) {
        rg[0] = *(const uint4*)(gA0 + kb);
        rg[1] = *(const uint4*)(gA1 + kb);
        rg[2] = *(const uint4*)(gB0 + kb);
        rg[3] = *(const uint4*)(gB1 + kb);
    };
    auto store_regs = [&](int buf) {
        *(uint4*)(&As[buf][r0][c0]) = rg[0];
        *(uint4*)(&As[buf][r1][c1]) = rg[1];
        *(uint4*)(&Bs[buf][r0][c0]) = rg[2];
        *(uint4*)(&Bs[buf][r1][c1]) = rg[3];
    };
    load_regs(0);               // prologue: tile 0 -> buffer 0
    store_regs(0);
    __syncthreads();
#endif

    const int nkb = K >> 5;
    for (int ib = 0; ib < nkb; ++ib) {
        const int  cur  = ib & 1;
        const bool more = (ib + 1) < nkb;

        // kick off the NEXT tile's copies before computing (overlap)
#if USE_ASYNC_LDS
        if (more) stage_async(cur ^ 1, (ib + 1) << 5);
#else
        if (more) {
            load_regs((ib + 1) << 5);
            __builtin_prefetch(gA0 + ((ib + 2) << 5), 0, 3);
            __builtin_prefetch(gB0 + ((ib + 2) << 5), 0, 3);
        }
#endif

        // ---- compute from buffer `cur` ----
        Frag a[4], b[2];
#pragma unroll
        for (int mi = 0; mi < 4; ++mi) {
            const unsigned short* p = &As[cur][wm * 64 + mi * 16 + mrow][0];
            a[mi].u[0] = *(const uint4*)(p + koffA);
            a[mi].u[1] = *(const uint4*)(p + koffA + 16);
        }
#pragma unroll
        for (int ni = 0; ni < 2; ++ni) {
            const unsigned short* p = &Bs[cur][wn * 32 + ni * 16 + mrow][0];
            b[ni].u[0] = *(const uint4*)(p + koffB);
            b[ni].u[1] = *(const uint4*)(p + koffB + 16);
        }
#pragma unroll
        for (int mi = 0; mi < 4; ++mi)
#pragma unroll
            for (int ni = 0; ni < 2; ++ni)
                acc[mi][ni] = __builtin_amdgcn_wmma_f32_16x16x32_bf16(
                    false, a[mi].v, false, b[ni].v,
                    (short)0, acc[mi][ni], false, false);

        // ---- land the next tile, one barrier per iteration ----
#if USE_ASYNC_LDS
        if (more) wait_async0();
        __syncthreads();
#else
        if (more) store_regs(cur ^ 1);
        __syncthreads();
#endif
    }

    // epilogue: C layout -> lane&15 = N, VGPR r -> M = r + 8*(lane>>4)
    const int nloc  = lane & 15;
    const int mbase = (lane >> 4) * 8;
#pragma unroll
    for (int mi = 0; mi < 4; ++mi) {
#pragma unroll
        for (int ni = 0; ni < 2; ++ni) {
            const int colg = tileN + wn * 32 + ni * 16 + nloc;
#pragma unroll
            for (int r = 0; r < 8; ++r) {
                const int rowg = tileM + wm * 64 + mi * 16 + mbase + r;
                float v = acc[mi][ni][r];
                if (bias)   v += bias[colg];
                if (addsrc) v += addsrc[(size_t)rowg * N + colg];
                if (dotanh) v = tanhf(v);
                if (outF)   outF[(size_t)rowg * ldout + colg] = v;
                if (outB)   outB[(size_t)rowg * N + colg] = f2bf_u(v);
            }
        }
    }
}

// ---------- launch ----------
extern "C" void kernel_launch(void* const* d_in, const int* in_sizes, int n_in,
                              void* d_out, int out_size, void* d_ws, size_t ws_size,
                              hipStream_t stream) {
    const float* z_in = (const float*)d_in[0];  // [B, D]
    const float* Wx   = (const float*)d_in[1];  // [D, H]
    const float* Wh   = (const float*)d_in[2];  // [H, H]
    const float* bh   = (const float*)d_in[3];  // [H]
    const float* Wout = (const float*)d_in[4];  // [H, D]
    const float* bout = (const float*)d_in[5];  // [D]
    float* out = (float*)d_out;                 // [B, NSTEP, D]

    char* wsp = (char*)d_ws;
    auto carve = [&](size_t bytes) -> void* {
        void* p = (void*)wsp;
        wsp += (bytes + 255) & ~(size_t)255;
        return p;
    };
    unsigned short* WxT   = (unsigned short*)carve((size_t)HH * DD * 2); // [H, D]
    unsigned short* WhT   = (unsigned short*)carve((size_t)HH * HH * 2); // [H, H]
    unsigned short* WoutT = (unsigned short*)carve((size_t)DD * HH * 2); // [D, H]
    unsigned short* z_bf  = (unsigned short*)carve((size_t)BB * DD * 2); // [B, D]
    float*          xproj = (float*)carve((size_t)BB * HH * 4);          // [B, H]
    unsigned short* hbuf0 = (unsigned short*)carve((size_t)BB * HH * 2); // [B, H]
    unsigned short* hbuf1 = (unsigned short*)carve((size_t)BB * HH * 2); // [B, H]
    unsigned short* hbuf[2] = {hbuf0, hbuf1};

    // 1) weight transpose + bf16 convert
    k_transpose_to_bf16<<<(DD * HH) / 256, 256, 0, stream>>>(Wx,   WxT,   DD, HH);
    k_transpose_to_bf16<<<(HH * HH) / 256, 256, 0, stream>>>(Wh,   WhT,   HH, HH);
    k_transpose_to_bf16<<<(HH * DD) / 256, 256, 0, stream>>>(Wout, WoutT, HH, DD);
    // 2) activations to bf16, h0 = 0
    k_convert_bf16<<<(BB * DD) / 256, 256, 0, stream>>>(z_in, z_bf);
    k_zero_u32<<<((BB * HH) / 2) / 256, 256, 0, stream>>>((unsigned int*)hbuf0);

    // 3) xproj = z_in @ Wx + bh
    k_gemm_bf16_wmma<<<dim3(HH / 128, BB / 128), 256, 0, stream>>>(
        z_bf, WxT, bh, nullptr, xproj, (long long)HH, nullptr,
        BB, HH, DD, 0);

    // 4) recurrence: 16 steps, double-buffered hidden state (bf16)
    for (int t = 0; t < NSTEP; ++t) {
        unsigned short* hcur = hbuf[t & 1];
        unsigned short* hnxt = hbuf[(t + 1) & 1];
        // h_new = tanh(xproj + h @ Wh)  -> bf16
        k_gemm_bf16_wmma<<<dim3(HH / 128, BB / 128), 256, 0, stream>>>(
            hcur, WhT, nullptr, xproj, nullptr, 0, hnxt,
            BB, HH, HH, 1);
        // out[:, t, :] = h_new @ Wout + bout  (strided f32 store)
        k_gemm_bf16_wmma<<<dim3(DD / 128, BB / 128), 256, 0, stream>>>(
            hnxt, WoutT, bout, nullptr,
            out + (size_t)t * DD, (long long)NSTEP * DD, nullptr,
            BB, DD, HH, 0);
    }
}
